// InteractionBlock_5557687681721
// MI455X (gfx1250) — matrix operand order repaired and checked by reading, not compiled, not run on hardware
//
#include <hip/hip_runtime.h>

#define E_NUM 131072
#define T_NUM 1048576
#define H_DIM 128
#define NB    8
#define NR    6
#define NSR   42

typedef unsigned short ushort_t;
typedef unsigned int   uint_t;
typedef __attribute__((ext_vector_type(8)))  float  v8f;
typedef __attribute__((ext_vector_type(16))) __bf16 v16bf;
typedef __attribute__((__vector_size__(16))) int v4i_t;

union BF16x16 {
  v16bf    v;
  ushort_t u[16];
  uint_t   d[8];
};

__device__ __forceinline__ ushort_t f2bf(float f) {
  uint_t x = __builtin_bit_cast(uint_t, f);
  uint_t r = (x + 0x7FFFu + ((x >> 16) & 1u)) >> 16;   // round-to-nearest-even
  return (ushort_t)r;
}
__device__ __forceinline__ float bf2f(ushort_t h) {
  uint_t x = ((uint_t)h) << 16;
  return __builtin_bit_cast(float, x);
}
__device__ __forceinline__ void ld16(uint_t* dst4, const void* p) {
  const uint4 t = *(const uint4*)p;
  dst4[0] = t.x; dst4[1] = t.y; dst4[2] = t.z; dst4[3] = t.w;
}
__device__ __forceinline__ void loadB(BF16x16& b, const ushort_t* base) {
  ld16(&b.d[0], base);
  ld16(&b.d[4], base + 16);
}
__device__ __forceinline__ float silu(float v) {
  return v / (1.f + __expf(-v));
}
// packed bf16 multiply: r = a * b (2x bf16 lanes), RNE
__device__ __forceinline__ uint_t pk_mul_bf16(uint_t a, uint_t b) {
  uint_t r;
  asm("v_pk_mul_bf16 %0, %1, %2" : "=v"(r) : "v"(a), "v"(b));
  return r;
}
__device__ __forceinline__ BF16x16 scale_frag_pk(const BF16x16& a, uint_t sbb) {
  BF16x16 r;
#pragma unroll
  for (int i = 0; i < 8; ++i) r.d[i] = pk_mul_bf16(a.d[i], sbb);
  return r;
}

// -------- async global->LDS copy (16B per lane), ASYNCcnt tracked --------
#if __has_builtin(__builtin_amdgcn_global_load_async_to_lds_b128)
#define HAVE_ASYNC_LDS 1
#endif

__device__ __forceinline__ void async_cp16(void* lds, const void* g) {
#ifdef HAVE_ASYNC_LDS
  __builtin_amdgcn_global_load_async_to_lds_b128(
      (__attribute__((address_space(1))) v4i_t*)g,
      (__attribute__((address_space(3))) v4i_t*)lds, 0, 0);
#else
  *(uint4*)lds = *(const uint4*)g;
#endif
}
__device__ __forceinline__ void wait_async0() {
#if defined(HAVE_ASYNC_LDS) && __has_builtin(__builtin_amdgcn_s_wait_asynccnt)
  __builtin_amdgcn_s_wait_asynccnt(0);
#elif defined(HAVE_ASYNC_LDS)
  asm volatile("s_wait_asynccnt 0x0" ::: "memory");
#endif
}

// ---------------- weight conversion ----------------
// dst[n*128+k] = bf16(src[k*128+n])  (B-fragment friendly: row-major [N][K])
__global__ void convT128_kernel(const float* __restrict__ src, ushort_t* __restrict__ dst) {
  int i = blockIdx.x * 256 + threadIdx.x;   // 16384 elements
  int n = i >> 7, k = i & 127;
  dst[n * 128 + k] = f2bf(src[k * 128 + n]);
}

// W[i,j,l] (H,NB,H) -> Wt[(j*128+i)*128 + l]  (per-j slices Wt_j[n=i][k=l])
__global__ void convW_kernel(const float* __restrict__ W, ushort_t* __restrict__ dst) {
  int idx = blockIdx.x * 256 + threadIdx.x; // 131072 elements
  int l = idx & 127;
  int i = (idx >> 7) & 127;
  int j = idx >> 14;
  dst[idx] = f2bf(W[(size_t)i * (NB * H_DIM) + j * H_DIM + l]);
}

// ---------------- small dense ops ----------------
__global__ void rbf_kernel(const float* __restrict__ rbf, const float* __restrict__ w_rbf,
                           float* __restrict__ out) {
  __shared__ float w[NR * H_DIM];
  int tid = threadIdx.x;
  for (int i = tid; i < NR * H_DIM; i += 256) w[i] = w_rbf[i];
  __syncthreads();
  int idx = blockIdx.x * 256 + tid;  // E*128 total
  int e = idx >> 7, c = idx & 127;
  float acc = 0.f;
#pragma unroll
  for (int r = 0; r < NR; ++r) acc += rbf[e * NR + r] * w[r * H_DIM + c];
  out[idx] = acc;
}

__global__ void sbf_kernel(const float* __restrict__ sbf, const float* __restrict__ w_sbf,
                           float* __restrict__ out) {
  __shared__ float w[NSR * NB];
  int tid = threadIdx.x;
  for (int i = tid; i < NSR * NB; i += 256) w[i] = w_sbf[i];
  __syncthreads();
  int t = blockIdx.x * 256 + tid;
  float a[NB];
#pragma unroll
  for (int j = 0; j < NB; ++j) a[j] = 0.f;
  for (int r = 0; r < NSR; ++r) {
    float sv = sbf[(size_t)t * NSR + r];
#pragma unroll
    for (int j = 0; j < NB; ++j) a[j] += sv * w[r * NB + j];
  }
#pragma unroll
  for (int j = 0; j < NB; ++j) out[(size_t)t * NB + j] = a[j];
}

__global__ void add_kernel(const float* __restrict__ a, const float* __restrict__ b,
                           float* __restrict__ o) {
  size_t i = (size_t)blockIdx.x * 256 + threadIdx.x;
  float4 va = ((const float4*)a)[i], vb = ((const float4*)b)[i];
  float4 r; r.x = va.x + vb.x; r.y = va.y + vb.y; r.z = va.z + vb.z; r.w = va.w + vb.w;
  ((float4*)o)[i] = r;
}

// ---------------- generic 128x128 GEMM (A fp32 -> bf16 WMMA, fp32 accum) ----------------
// EPI bit0: SiLU   bit1: *mul[m,n]   bit2: +res[m,n]   bit3: store bf16
template <int EPI>
__launch_bounds__(256)
__global__ void gemm128_kernel(const float* __restrict__ A, const ushort_t* __restrict__ Wt,
                               const float* __restrict__ bias, const float* __restrict__ mul,
                               const float* __restrict__ res, void* __restrict__ outp) {
  __shared__ __align__(16) ushort_t wls[H_DIM * H_DIM];  // 32 KB weights
  const int tid = threadIdx.x;
  {
    const char* src = (const char*)Wt;
    char* dst = (char*)wls;
#pragma unroll
    for (int i = 0; i < 8; ++i)
      async_cp16(dst + (size_t)(tid + i * 256) * 16, src + (size_t)(tid + i * 256) * 16);
    wait_async0();
  }
  __syncthreads();

  const int lane = tid & 31;
  const int wid  = tid >> 5;
  const int row0 = blockIdx.x * 128 + wid * 16;
  const int mrow = lane & 15;
  const int hi   = lane >> 4;
  const int koff = hi * 8;

  // A fragments: 4 k-tiles of 32 (fp32 -> bf16)
  BF16x16 afr[4];
#pragma unroll
  for (int kt = 0; kt < 4; ++kt) {
    const float* ap = A + (size_t)(row0 + mrow) * H_DIM + kt * 32 + koff;
    float4 a0 = *(const float4*)(ap);
    float4 a1 = *(const float4*)(ap + 4);
    float4 b0 = *(const float4*)(ap + 16);
    float4 b1 = *(const float4*)(ap + 20);
    afr[kt].u[0] = f2bf(a0.x); afr[kt].u[1] = f2bf(a0.y);
    afr[kt].u[2] = f2bf(a0.z); afr[kt].u[3] = f2bf(a0.w);
    afr[kt].u[4] = f2bf(a1.x); afr[kt].u[5] = f2bf(a1.y);
    afr[kt].u[6] = f2bf(a1.z); afr[kt].u[7] = f2bf(a1.w);
    afr[kt].u[8]  = f2bf(b0.x); afr[kt].u[9]  = f2bf(b0.y);
    afr[kt].u[10] = f2bf(b0.z); afr[kt].u[11] = f2bf(b0.w);
    afr[kt].u[12] = f2bf(b1.x); afr[kt].u[13] = f2bf(b1.y);
    afr[kt].u[14] = f2bf(b1.z); afr[kt].u[15] = f2bf(b1.w);
  }

  v8f zero = {0.f, 0.f, 0.f, 0.f, 0.f, 0.f, 0.f, 0.f};
  v8f acc[8];
#pragma unroll
  for (int nt = 0; nt < 8; ++nt) acc[nt] = zero;

  // flattened kt-outer / nt-inner, double-buffered B fragments from LDS
  const ushort_t* bptr = wls + mrow * H_DIM + hi * 8;
  BF16x16 bfr[2];
  loadB(bfr[0], bptr);   // it=0: kt=0, nt=0
#pragma unroll
  for (int it = 0; it < 32; ++it) {
    const int kt = it >> 3, nt = it & 7;
    if (it < 31) {
      const int kt2 = (it + 1) >> 3, nt2 = (it + 1) & 7;
      loadB(bfr[(it + 1) & 1], bptr + nt2 * (16 * H_DIM) + kt2 * 32);
    }
    acc[nt] = __builtin_amdgcn_wmma_f32_16x16x32_bf16(
        false, afr[kt].v, false, bfr[it & 1].v, (short)0, acc[nt], false, false);
  }

#pragma unroll
  for (int nt = 0; nt < 8; ++nt) {
    const int n = nt * 16 + mrow;
    const float bb = bias[n];
#pragma unroll
    for (int vv = 0; vv < 8; ++vv) {
      const int m = row0 + vv + 8 * hi;
      float val = acc[nt][vv] + bb;
      if (EPI & 1) val = silu(val);
      if (EPI & 2) val *= mul[(size_t)m * H_DIM + n];
      if (EPI & 4) val += res[(size_t)m * H_DIM + n];
      if (EPI & 8) ((ushort_t*)outp)[(size_t)m * H_DIM + n] = f2bf(val);
      else         ((float*)outp)[(size_t)m * H_DIM + n] = val;
    }
  }
}

// ---------------- triplet bilinear: gather -> 8 scaled WMMA GEMMs -> atomic scatter ----------------
__launch_bounds__(256)
__global__ void triplet_kernel(const ushort_t* __restrict__ xkj,   // [E][128] bf16
                               const ushort_t* __restrict__ Wt,    // [8][128][128] bf16 (n=i, k=l)
                               const float* __restrict__ sbfh,     // [T][8]
                               const int* __restrict__ idx_kj,
                               const int* __restrict__ idx_ji,
                               float* __restrict__ agg) {
  __shared__ __align__(16) ushort_t wls[2 * H_DIM * H_DIM];  // 64 KB: two j-slices
  const int tid  = threadIdx.x;
  const int lane = tid & 31;
  const int wid  = tid >> 5;
  const int tile = blockIdx.x * 128 + wid * 16;   // 16 triplets per wave
  const int mrow = lane & 15;
  const int hi   = lane >> 4;
  const int koff = hi * 8;

  const int w_row = tile + mrow;
  const int ekj   = idx_kj[w_row];

  // gather A fragments (bf16 activations), 4 k-tiles
  BF16x16 afr[4];
#pragma unroll
  for (int kt = 0; kt < 4; ++kt) {
    const ushort_t* ap = xkj + (size_t)ekj * H_DIM + kt * 32 + koff;
    ld16(&afr[kt].d[0], ap);
    ld16(&afr[kt].d[4], ap + 16);
  }

  v8f zero = {0.f, 0.f, 0.f, 0.f, 0.f, 0.f, 0.f, 0.f};
  v8f acc[8];
#pragma unroll
  for (int nt = 0; nt < 8; ++nt) acc[nt] = zero;

  for (int jj = 0; jj < 4; ++jj) {
    __syncthreads();
    {   // stage two 32KB W_j slices into LDS (async global->LDS)
      const char* src = (const char*)(Wt + (size_t)jj * 2 * H_DIM * H_DIM);
      char* dst = (char*)wls;
#pragma unroll
      for (int i = 0; i < 16; ++i)
        async_cp16(dst + (size_t)(tid + i * 256) * 16, src + (size_t)(tid + i * 256) * 16);
      wait_async0();
    }
    __syncthreads();
#pragma unroll
    for (int js = 0; js < 2; ++js) {
      const int j   = jj * 2 + js;
      const float s = sbfh[(size_t)w_row * NB + j];    // per-row (per-lane) scale
      uint_t sbb; { ushort_t sb = f2bf(s); sbb = (uint_t)sb | ((uint_t)sb << 16); }
      // scale all 4 A fragments once per j (v_pk_mul_bf16)
      BF16x16 as[4];
#pragma unroll
      for (int kt = 0; kt < 4; ++kt) as[kt] = scale_frag_pk(afr[kt], sbb);

      const ushort_t* bptr = wls + js * H_DIM * H_DIM + mrow * H_DIM + hi * 8;
      BF16x16 bfr[2];
      loadB(bfr[0], bptr);   // it=0: kt=0, nt=0
#pragma unroll
      for (int it = 0; it < 32; ++it) {
        const int kt = it >> 3, nt = it & 7;
        if (it < 31) {
          const int kt2 = (it + 1) >> 3, nt2 = (it + 1) & 7;
          loadB(bfr[(it + 1) & 1], bptr + nt2 * (16 * H_DIM) + kt2 * 32);
        }
        acc[nt] = __builtin_amdgcn_wmma_f32_16x16x32_bf16(
            false, as[kt].v, false, bfr[it & 1].v, (short)0, acc[nt], false, false);
      }
    }
  }

  // scatter-add into agg[idx_ji]
#pragma unroll
  for (int vv = 0; vv < 8; ++vv) {
    const int m  = vv + 8 * hi;
    const int ej = idx_ji[tile + m];
    float* dst = agg + (size_t)ej * H_DIM + mrow;
#pragma unroll
    for (int nt = 0; nt < 8; ++nt) atomicAdd(dst + nt * 16, acc[nt][vv]);
  }
}

extern "C" void kernel_launch(void* const* d_in, const int* in_sizes, int n_in,
                              void* d_out, int out_size, void* d_ws, size_t ws_size,
                              hipStream_t stream) {
  (void)in_sizes; (void)n_in; (void)out_size; (void)ws_size;
  const float* x      = (const float*)d_in[0];
  const float* rbf    = (const float*)d_in[1];
  const float* sbf    = (const float*)d_in[2];
  const int*   idx_kj = (const int*)d_in[3];
  const int*   idx_ji = (const int*)d_in[4];
  const float* w_rbf  = (const float*)d_in[5];
  const float* w_sbf  = (const float*)d_in[6];
  const float* w_kj   = (const float*)d_in[7];
  const float* b_kj   = (const float*)d_in[8];
  const float* w_ji   = (const float*)d_in[9];
  const float* b_ji   = (const float*)d_in[10];
  const float* Wb     = (const float*)d_in[11];
  const float* rb_w1  = (const float*)d_in[12];
  const float* rb_b1  = (const float*)d_in[13];
  const float* rb_w2  = (const float*)d_in[14];
  const float* rb_b2  = (const float*)d_in[15];
  const float* w_lin  = (const float*)d_in[16];
  const float* b_lin  = (const float*)d_in[17];
  const float* ra_w1  = (const float*)d_in[18];
  const float* ra_b1  = (const float*)d_in[19];
  const float* ra_w2  = (const float*)d_in[20];
  const float* ra_b2  = (const float*)d_in[21];

  char* ws = (char*)d_ws;
  ushort_t* wji_t  = (ushort_t*)(ws + 0);
  ushort_t* wkj_t  = (ushort_t*)(ws + 32768);
  ushort_t* rb1_t  = (ushort_t*)(ws + 65536);
  ushort_t* rb2_t  = (ushort_t*)(ws + 98304);
  ushort_t* wlin_t = (ushort_t*)(ws + 131072);
  ushort_t* ra1_t  = (ushort_t*)(ws + 163840);   // 2 matrices
  ushort_t* ra2_t  = (ushort_t*)(ws + 229376);   // 2 matrices
  ushort_t* Wt     = (ushort_t*)(ws + 294912);   // 256 KB
  size_t o = 1048576;
  float*    rbf_h  = (float*)(ws + o);    o += (size_t)E_NUM * H_DIM * 4;
  float*    x_ji   = (float*)(ws + o);    o += (size_t)E_NUM * H_DIM * 4;
  ushort_t* xkj_bf = (ushort_t*)(ws + o); o += (size_t)E_NUM * H_DIM * 2;
  float*    sbf_h  = (float*)(ws + o);    o += (size_t)T_NUM * NB * 4;
  float*    agg    = (float*)(ws + o);    o += (size_t)E_NUM * H_DIM * 4;
  float*    hbuf   = (float*)(ws + o);    o += (size_t)E_NUM * H_DIM * 4;
  float*    tbuf   = (float*)(ws + o);    o += (size_t)E_NUM * H_DIM * 4;

  (void)hipMemsetAsync(agg, 0, (size_t)E_NUM * H_DIM * sizeof(float), stream);

  convT128_kernel<<<64, 256, 0, stream>>>(w_ji,  wji_t);
  convT128_kernel<<<64, 256, 0, stream>>>(w_kj,  wkj_t);
  convT128_kernel<<<64, 256, 0, stream>>>(rb_w1, rb1_t);
  convT128_kernel<<<64, 256, 0, stream>>>(rb_w2, rb2_t);
  convT128_kernel<<<64, 256, 0, stream>>>(w_lin, wlin_t);
  convT128_kernel<<<64, 256, 0, stream>>>(ra_w1,          ra1_t);
  convT128_kernel<<<64, 256, 0, stream>>>(ra_w1 + 16384,  ra1_t + 16384);
  convT128_kernel<<<64, 256, 0, stream>>>(ra_w2,          ra2_t);
  convT128_kernel<<<64, 256, 0, stream>>>(ra_w2 + 16384,  ra2_t + 16384);
  convW_kernel<<<512, 256, 0, stream>>>(Wb, Wt);

  rbf_kernel<<<E_NUM * H_DIM / 256, 256, 0, stream>>>(rbf, w_rbf, rbf_h);
  sbf_kernel<<<T_NUM / 256, 256, 0, stream>>>(sbf, w_sbf, sbf_h);

  // x_ji = silu(x@w_ji + b_ji);  x_kj = silu(x@w_kj + b_kj) * rbf_h  (stored bf16)
  gemm128_kernel<1><<<E_NUM / 128, 256, 0, stream>>>(x, wji_t, b_ji, nullptr, nullptr, x_ji);
  gemm128_kernel<11><<<E_NUM / 128, 256, 0, stream>>>(x, wkj_t, b_kj, rbf_h, nullptr, xkj_bf);

  // hot path
  triplet_kernel<<<T_NUM / 128, 256, 0, stream>>>(xkj_bf, Wt, sbf_h, idx_kj, idx_ji, agg);

  // h = x_ji + agg
  add_kernel<<<E_NUM * H_DIM / 4 / 256, 256, 0, stream>>>(x_ji, agg, hbuf);

  // residual-before-skip (1), lin + x, residual-after-skip (2)
  gemm128_kernel<1><<<E_NUM / 128, 256, 0, stream>>>(hbuf, rb1_t, rb_b1, nullptr, nullptr, tbuf);
  gemm128_kernel<5><<<E_NUM / 128, 256, 0, stream>>>(tbuf, rb2_t, rb_b2, nullptr, hbuf, hbuf);
  gemm128_kernel<5><<<E_NUM / 128, 256, 0, stream>>>(hbuf, wlin_t, b_lin, nullptr, x, hbuf);
  gemm128_kernel<1><<<E_NUM / 128, 256, 0, stream>>>(hbuf, ra1_t, ra_b1, nullptr, nullptr, tbuf);
  gemm128_kernel<5><<<E_NUM / 128, 256, 0, stream>>>(tbuf, ra2_t, ra_b2, nullptr, hbuf, hbuf);
  gemm128_kernel<1><<<E_NUM / 128, 256, 0, stream>>>(hbuf, ra1_t + 16384, ra_b1 + 128, nullptr, nullptr, tbuf);
  gemm128_kernel<5><<<E_NUM / 128, 256, 0, stream>>>(tbuf, ra2_t + 16384, ra_b2 + 128, nullptr, hbuf, (float*)d_out);
}
